// LSTMLayer_3066606649427
// MI455X (gfx1250) — matrix operand order repaired and checked by reading, not compiled, not run on hardware
//
#include <hip/hip_runtime.h>

// LSTM: B=256, T=512, F=64, U=256.
// 16 persistent workgroups, each owns 16 batch rows for the whole time loop.
// Weights are packed once into f16 column-major Wt[n][k] in d_ws and streamed
// from L2 every step as contiguous per-lane WMMA B-fragments (line-perfect).
// An opaque zero-increment on the weight pointers stops LICM from hoisting
// the loads (which previously recreated a 160-VGPR live range and spilled).
// h/c/z state lives in LDS; f16 WMMA with f32 accumulation.

#define B_  256
#define T_  512
#define F_  64
#define U_  256
#define G_  1024          // 4*U
#define K_  320           // F + U
#define KSTEPS 10         // K_/32
#define AS 328            // A-staging row stride (halves), 16B aligned
#define ZS 1025           // z row stride (floats), odd -> bank spread
#define CS 257            // c row stride (floats)

typedef __attribute__((ext_vector_type(16))) _Float16 v16h;
typedef __attribute__((ext_vector_type(8)))  _Float16 v8h;
typedef __attribute__((ext_vector_type(8)))  float    v8f;

__device__ __forceinline__ float sigmf(float x)     { return 1.0f / (1.0f + __expf(-x)); }
__device__ __forceinline__ float tanh_fast(float x) { return 1.0f - 2.0f / (__expf(2.0f * x) + 1.0f); }

// ---- Kernel 1: pack [Wx; Wr] (f32, row-major KxN) -> Wt (f16, column-major NxK).
// Wt[n*320 + k]; column stride 640B (10 cache lines), so B-fragment loads are
// 32B-aligned contiguous runs and lane pairs (n, n+16) consume whole 64B lines.
__global__ void pack_weights(const float* __restrict__ wk,
                             const float* __restrict__ wr,
                             _Float16* __restrict__ wt) {
  const int idx = blockIdx.x * blockDim.x + threadIdx.x;   // 0 .. 1024*320-1
  const int n = idx / K_;
  const int k = idx - n * K_;
  const float v = (k < F_) ? wk[k * G_ + n] : wr[(k - F_) * G_ + n];
  wt[idx] = (_Float16)v;
}

// ---- Kernel 2: persistent LSTM over T timesteps.
__global__ __launch_bounds__(1024) void lstm_persistent(
    const float* __restrict__ x,     // [B,T,F]
    const _Float16* __restrict__ wt, // [N=1024][K=320] f16 column-major
    const float* __restrict__ bias,  // [4U]
    const float* __restrict__ h0,    // [B,U]
    const float* __restrict__ c0,    // [B,U]
    float* __restrict__ out)         // [B,U]
{
  __shared__ __align__(16) _Float16 Alds[16 * AS]; // [m][k]: k<64 = x_t, k>=64 = h (f16)
  __shared__ float Zlds[16 * ZS];                  // gate pre-activations (f32)
  __shared__ float Cst [16 * CS];                  // cell state (f32)

  const int tid     = threadIdx.x;
  const int lane    = tid & 31;
  const int wave    = tid >> 5;        // 0..31
  const int rowbase = blockIdx.x * 16; // batch rows owned by this WG

  // This wave covers N columns [n0, n0+32) as two 16-wide WMMA tiles.
  const int n0 = wave * 32;
  // B-fragment per-lane base: column n = tile_n0 + (lane&15),
  // K-run start = (lane>=16 ? 16 : 0)  (contiguous K order, same as A).
  const _Float16* bbase0 = wt + (size_t)(n0 + (lane & 15)) * K_ + ((lane >> 4) * 16);
  const _Float16* bbase1 = bbase0 + (size_t)16 * K_;

  // Opaque zero: formally loop-variant weight pointers so LICM cannot hoist
  // the B-fragment loads out of the time loop (they must stream from L2).
  int zero = 0;
  asm volatile("" : "+s"(zero));

  // ---- elementwise mapping: thread owns hidden unit u for rows m0,m0+4,m0+8,m0+12
  const int u  = tid & 255;
  const int m0 = tid >> 8;
  const float b_i = bias[u];
  const float b_f = bias[U_ + u];
  const float b_o = bias[2 * U_ + u];
  const float b_g = bias[3 * U_ + u];

  // ---- initialize LDS with x_0, h0 (f16) and c0 (f32)
  {
    const int m = tid >> 6, f = tid & 63;
    Alds[m * AS + f] = (_Float16)x[(long)(rowbase + m) * T_ * F_ + f];
  }
#pragma unroll
  for (int r = 0; r < 4; ++r) {
    const int m = m0 + 4 * r;
    Alds[m * AS + F_ + u] = (_Float16)h0[(rowbase + m) * U_ + u];
    Cst[m * CS + u]       = c0[(rowbase + m) * U_ + u];
  }
  __syncthreads();

  // A-fragment addressing (16-bit A 16x32 layout, contiguous K to match B):
  // lane<16 : row=lane,    halves K = ks*32 + {0..15}
  // lane>=16: row=lane-16, halves K = ks*32 + {16..31}
  const _Float16* aptr = &Alds[(lane & 15) * AS + ((lane >> 4) * 16)];
  const int zn = n0 + (lane & 15);
  const int zm = (lane >> 4) * 8;

  for (int t = 0; t < T_; ++t) {
    v8f acc0 = {};
    v8f acc1 = {};
#pragma unroll
    for (int ks = 0; ks < KSTEPS; ++ks) {
      const v8h lo = *(const v8h*)(aptr + ks * 32);
      const v8h hi = *(const v8h*)(aptr + ks * 32 + 8);
      const v16h a = __builtin_shufflevector(lo, hi,
          0,1,2,3,4,5,6,7,8,9,10,11,12,13,14,15);
      // Contiguous 32B per-lane B-fragments, streamed from L2 each step.
      const v16h b0 = *(const v16h*)(bbase0 + ks * 32);
      const v16h b1 = *(const v16h*)(bbase1 + ks * 32);
      acc0 = __builtin_amdgcn_wmma_f32_16x16x32_f16(false, a, false, b0,
                                                    (short)0, acc0, false, false);
      acc1 = __builtin_amdgcn_wmma_f32_16x16x32_f16(false, a, false, b1,
                                                    (short)0, acc1, false, false);
    }
    // scatter z tiles to LDS (C/D layout: VGPR v -> row zm+v, col = lane&15)
#pragma unroll
    for (int v = 0; v < 8; ++v) {
      Zlds[(zm + v) * ZS + zn]      = acc0[v];
      Zlds[(zm + v) * ZS + zn + 16] = acc1[v];
    }
    __syncthreads();

    // gate math + state update (f32)
#pragma unroll
    for (int r = 0; r < 4; ++r) {
      const int m = m0 + 4 * r;
      const float zi = Zlds[m * ZS + u]           + b_i;
      const float zf = Zlds[m * ZS + U_ + u]      + b_f;
      const float zo = Zlds[m * ZS + 2 * U_ + u]  + b_o;
      const float zg = Zlds[m * ZS + 3 * U_ + u]  + b_g;
      const float ig = sigmf(zi);
      const float fg = sigmf(zf);
      const float og = sigmf(zo);
      const float gg = tanh_fast(zg);
      const float c  = fg * Cst[m * CS + u] + ig * gg;
      Cst[m * CS + u] = c;
      const float h = og * tanh_fast(c);
      Alds[m * AS + F_ + u] = (_Float16)h;          // h_t feeds next step's A
      if (t == T_ - 1) out[(rowbase + m) * U_ + u] = h;
    }
    // stage x_{t+1} into A, prefetch x_{t+2} (x is 33.5MB, L2-resident)
    if (t + 1 < T_) {
      const int m = tid >> 6, f = tid & 63;
      const long base = (long)(rowbase + m) * T_ * F_;
      Alds[m * AS + f] = (_Float16)x[base + (t + 1) * F_ + f];
      if (t + 2 < T_) __builtin_prefetch(&x[base + (t + 2) * F_ + f], 0, 1);
    }
    __syncthreads();

    // opaque (always zero) advance -> loads cannot be hoisted, no spills
    bbase0 += zero;
    bbase1 += zero;
  }
}

extern "C" void kernel_launch(void* const* d_in, const int* in_sizes, int n_in,
                              void* d_out, int out_size, void* d_ws, size_t ws_size,
                              hipStream_t stream) {
  const float* x  = (const float*)d_in[0];
  const float* wk = (const float*)d_in[1];
  const float* wr = (const float*)d_in[2];
  const float* bs = (const float*)d_in[3];
  const float* h0 = (const float*)d_in[4];
  const float* c0 = (const float*)d_in[5];
  float* out = (float*)d_out;

  _Float16* wt = (_Float16*)d_ws;   // 1024*320*2 = 640KB packed weights

  pack_weights<<<(G_ * K_) / 256, 256, 0, stream>>>(wk, wr, wt);
  lstm_persistent<<<B_ / 16, 1024, 0, stream>>>(x, wt, bs, h0, c0, out);
}
